// GraphEmb_56530359550353
// MI455X (gfx1250) — matrix operand order repaired and checked by reading, not compiled, not run on hardware
//
#include <hip/hip_runtime.h>
#include <hip/hip_bf16.h>
#include <math.h>

typedef __attribute__((ext_vector_type(16))) _Float16 v16h;
typedef __attribute__((ext_vector_type(8)))  float    v8f;

#define IN_DIM   256
#define SDIM     128
#define ROWS_PB  128     // rows per workgroup (8 waves x 16 rows)
#define NCTILES  9       // 8 output col tiles + 1 gate tile
#define LPAD     132     // LDS row stride (floats), avoids half-wave bank conflict

// ---------------------------------------------------------------------------
// Zero the output (harness poisons d_out with 0xAA; we accumulate atomically).
// ---------------------------------------------------------------------------
__global__ void zero_out_kernel(float* __restrict__ out, long long n) {
    long long i = (long long)blockIdx.x * blockDim.x + threadIdx.x;
    if (i < n) out[i] = 0.0f;
}

// ---------------------------------------------------------------------------
// Pre-swizzle W_f (256x128 f32) + W_g (256x1 f32) into f16 B-fragments.
// Layout: wt[((kc*9 + c)*32 + lane)*16 + j]
//   c in 0..7 : col = c*16 + (lane&15), k = kc*32 + (lane<16?0:8) + j + (j<8?0:8)  -> W_f
//   c == 8    : gate tile; only col 0 (lane&15 == 0) holds W_g, rest 0
// This matches the dense 16x16x32 f16 B layout: lane = N, lane-half 0 = K 0-7,16-23, half 1 = K 8-15,24-31,
// 16 packed halves per lane -> one contiguous 32B load per fragment in the main loop.
// ---------------------------------------------------------------------------
__global__ void prep_w_kernel(const float* __restrict__ Wf,
                              const float* __restrict__ Wg,
                              _Float16* __restrict__ wt) {
    int t = blockIdx.x * blockDim.x + threadIdx.x;   // one thread per (kc,c,lane)
    if (t >= 8 * NCTILES * 32) return;
    int lane = t & 31;
    int c    = (t >> 5) % NCTILES;
    int kc   = (t >> 5) / NCTILES;
    int kBase = kc * 32 + ((lane < 16) ? 0 : 8);
    int col   = c * 16 + (lane & 15);

    _Float16 vals[16];
#pragma unroll
    for (int j = 0; j < 16; ++j) {
        int k = kBase + j + ((j < 8) ? 0 : 8);
        float v;
        if (c < 8)                 v = Wf[k * SDIM + col];
        else if ((lane & 15) == 0) v = Wg[k];
        else                       v = 0.0f;
        vals[j] = (_Float16)v;
    }
    v16h* dst = (v16h*)(wt + (size_t)t * 16);
    *dst = *(const v16h*)vals;
}

// ---------------------------------------------------------------------------
// Fused: h@[W_f | W_g] via WMMA f16 -> gate -> bias -> LDS tile ->
// sorted-run segment reduction -> float atomics into out[65536,128].
// ---------------------------------------------------------------------------
__global__ __launch_bounds__(256)
void fused_gemm_gate_segsum(const float* __restrict__ h,
                            const int*   __restrict__ batch,
                            const float* __restrict__ bf,
                            const float* __restrict__ bg,
                            const _Float16* __restrict__ wt,
                            float* __restrict__ out,
                            int N) {
    __shared__ float lout[ROWS_PB * LPAD];
    __shared__ int   sbatch[ROWS_PB];

    const int tid  = threadIdx.x;
    const int wave = tid >> 5;
    const int lane = tid & 31;
    const long long blockRow = (long long)blockIdx.x * ROWS_PB;

    // batch ids for this block (invalid tail rows contribute 0, id is harmless)
    if (tid < ROWS_PB) {
        long long r = blockRow + tid;
        sbatch[tid] = (r < N) ? batch[r] : batch[N - 1];
    }

    // ---- A-fragment addressing (16-bit A 16x32 layout) ----
    // lanes 0-15: M=lane,   K chunks {0..7} and {16..23}
    // lanes 16-31: M=lane-16, K chunks {8..15} and {24..31}
    const int  rowL  = lane & 15;
    long long  rowG  = blockRow + wave * 16 + rowL;
    long long  rowC  = (rowG < N) ? rowG : (long long)(N - 1);
    const float* hrow = h + rowC * IN_DIM;
    const int  aBase = (lane < 16) ? 0 : 8;

    v8f zero8 = {};
    v8f acc[NCTILES];
#pragma unroll
    for (int c = 0; c < NCTILES; ++c) acc[c] = zero8;

    const v16h* wtv = (const v16h*)wt;

    for (int kc = 0; kc < 8; ++kc) {
        const float* pa = hrow + kc * 32 + aBase;
        float4 f0 = *(const float4*)(pa);
        float4 f1 = *(const float4*)(pa + 4);
        float4 f2 = *(const float4*)(pa + 16);
        float4 f3 = *(const float4*)(pa + 20);
        v16h a;
        a[0]  = (_Float16)f0.x; a[1]  = (_Float16)f0.y;
        a[2]  = (_Float16)f0.z; a[3]  = (_Float16)f0.w;
        a[4]  = (_Float16)f1.x; a[5]  = (_Float16)f1.y;
        a[6]  = (_Float16)f1.z; a[7]  = (_Float16)f1.w;
        a[8]  = (_Float16)f2.x; a[9]  = (_Float16)f2.y;
        a[10] = (_Float16)f2.z; a[11] = (_Float16)f2.w;
        a[12] = (_Float16)f3.x; a[13] = (_Float16)f3.y;
        a[14] = (_Float16)f3.z; a[15] = (_Float16)f3.w;

#pragma unroll
        for (int c = 0; c < NCTILES; ++c) {
            v16h b = wtv[(kc * NCTILES + c) * 32 + lane];
            acc[c] = __builtin_amdgcn_wmma_f32_16x16x32_f16(
                false, a, false, b, (short)0, acc[c], false, false);
        }
    }

    // ---- gate: tile 8, column 0 lives in lanes 0 (rows i) and 16 (rows i+8) ----
    const float bgv = bg[0];
    const int   srcLane = lane & 16;                 // broadcast source per half-wave
    const int   rBase   = wave * 16 + ((lane < 16) ? 0 : 8);
    float gate[8];
#pragma unroll
    for (int i = 0; i < 8; ++i) {
        float d  = acc[8][i];
        float gs = 1.0f / (1.0f + __expf(-(d + bgv)));
        gate[i]  = __shfl(gs, srcLane, 32);
        if (blockRow + rBase + i >= N) gate[i] = 0.0f;   // tail rows -> 0
    }

    // ---- epilogue: (acc + b_f) * gate -> LDS tile ----
    const int colLane = lane & 15;
#pragma unroll
    for (int c = 0; c < 8; ++c) {
        int col = c * 16 + colLane;
        float bfv = bf[col];
#pragma unroll
        for (int i = 0; i < 8; ++i) {
            int r = rBase + i;
            lout[r * LPAD + col] = (acc[c][i] + bfv) * gate[i];
        }
    }
    __syncthreads();

    // ---- sorted-run reduction: thread = (column, half); flush on id change ----
    const int col = tid & 127;
    const int r0  = (tid >> 7) * 64;
    int   cur = sbatch[r0];
    float sum = 0.0f;
    for (int r = r0; r < r0 + 64; ++r) {
        int s = sbatch[r];
        float v = lout[r * LPAD + col];
        if (s != cur) {
            unsafeAtomicAdd(&out[(size_t)cur * SDIM + col], sum);
            sum = 0.0f;
            cur = s;
        }
        sum += v;
    }
    unsafeAtomicAdd(&out[(size_t)cur * SDIM + col], sum);
}

// ---------------------------------------------------------------------------
extern "C" void kernel_launch(void* const* d_in, const int* in_sizes, int n_in,
                              void* d_out, int out_size, void* d_ws, size_t ws_size,
                              hipStream_t stream) {
    const float* h     = (const float*)d_in[0];
    const int*   batch = (const int*)  d_in[1];
    // d_in[2] = num_segments (implied by out_size)
    const float* Wf    = (const float*)d_in[3];
    const float* bf    = (const float*)d_in[4];
    const float* Wg    = (const float*)d_in[5];
    const float* bg    = (const float*)d_in[6];
    float* out = (float*)d_out;
    _Float16* wt = (_Float16*)d_ws;          // 8*9*32*16 halves = 73,728 bytes

    const int N = in_sizes[0] / IN_DIM;

    long long outN = (long long)out_size;
    zero_out_kernel<<<(int)((outN + 255) / 256), 256, 0, stream>>>(out, outN);

    const int prepThreads = 8 * NCTILES * 32;   // 2304
    prep_w_kernel<<<(prepThreads + 255) / 256, 256, 0, stream>>>(Wf, Wg, wt);

    const int nBlocks = (N + ROWS_PB - 1) / ROWS_PB;
    fused_gemm_gate_segsum<<<nBlocks, 256, 0, stream>>>(h, batch, bf, bg, wt, out, N);
}
